// ModularAdditionModel_74122545594780
// MI455X (gfx1250) — matrix operand order repaired and checked by reading, not compiled, not run on hardware
//
#include <hip/hip_runtime.h>

// Problem constants (reference: P=32000, D=128, M=2, B=4096)
#define P_DIM 32000
#define D_DIM 128
#define B_DIM 4096

typedef __attribute__((ext_vector_type(16))) __bf16 v16bf;
typedef __attribute__((ext_vector_type(8)))  __bf16 v8bf;
typedef __attribute__((ext_vector_type(8)))  float  v8f;
typedef __attribute__((ext_vector_type(4)))  float  v4f;

// f32 -> bf16 round-to-nearest-even via integer math
__device__ __forceinline__ unsigned short f32_to_bf16_rne(float f) {
    union { float f; unsigned int u; } c; c.f = f;
    unsigned int u = c.u;
    u += 0x7FFFu + ((u >> 16) & 1u);
    return (unsigned short)(u >> 16);
}

// ---------------------------------------------------------------------------
// Kernel 1: V[P,D] f32 -> Vbf[P,D] bf16  (one-time 16.4MB read / 8.2MB write)
// ---------------------------------------------------------------------------
__global__ void __launch_bounds__(256) k_convert_V(const float* __restrict__ V,
                                                   unsigned short* __restrict__ Vbf) {
    int idx = (blockIdx.x * 256 + threadIdx.x) * 4;
    float4 v = *(const float4*)(V + idx);
    ushort4 o;
    o.x = f32_to_bf16_rne(v.x);
    o.y = f32_to_bf16_rne(v.y);
    o.z = f32_to_bf16_rne(v.z);
    o.w = f32_to_bf16_rne(v.w);
    *(ushort4*)(Vbf + idx) = o;
}

// ---------------------------------------------------------------------------
// Kernel 2: hiddenT[B,D] bf16, hiddenT[b][d] = relu(W[d,x[b,0]] + W[d,x[b,1]])
// ---------------------------------------------------------------------------
__global__ void __launch_bounds__(256) k_build_hidden(const int* __restrict__ x,
                                                      const float* __restrict__ W,
                                                      unsigned short* __restrict__ hT) {
    int idx = blockIdx.x * 256 + threadIdx.x;   // [0, B*D)
    int b = idx >> 7;
    int d = idx & 127;
    int x0 = x[b * 2 + 0];
    int x1 = x[b * 2 + 1];
    const float* wrow = W + (size_t)d * P_DIM;
    float h = wrow[x0] + wrow[x1];
    h = h > 0.0f ? h : 0.0f;
    hT[idx] = f32_to_bf16_rne(h);
}

// ---------------------------------------------------------------------------
// Kernel 3: out[b,p] = sum_d Vbf[p,d] * hiddenT[b,d] via v_wmma_f32_16x16x32_bf16
//
// Block = 256 thr = 8 waves = 8 p-tiles (128 p rows) x 128 b columns.
//  * B strip (128 rows x 128 d bf16 = 32 KB) staged ONCE per block into LDS
//    with CDNA5 global_load_async_to_lds_b128 (ASYNCcnt) -> kills the 8x
//    redundant per-wave L2 B reads.
//  * Each wave: A tile (16p x 128d) held in regs (4 x v16bf, reused across all
//    8 N-tiles), 8 f32 accumulators, 32 WMMAs, B fragments via ds_load_b128.
//  * Output: 32 contiguous bytes/lane, non-temporal (524 MB write-once stream
//    must not evict L2-resident Vbf/hiddenT).
// Grid = (2000/8, 4096/128) = (250, 32).
//
// Per-lane (lane = r + 16*hi) fragment layout, per ISA §7.12.2:
//   A 16x32 bf16 : row M=r; v[0..7]=K(kc*32+hi*8+0..7), v[8..15]=+16
//   B 32x16 bf16 : col N=r; v[0..7]=K(kc*32+hi*8+0..7), v[8..15]=+16
//   C/D 16x16 f32: elem v -> D[m=hi*8+v][n=r]
// ---------------------------------------------------------------------------
__global__ void __launch_bounds__(256) k_gemm_wmma(const unsigned short* __restrict__ Vbf,
                                                   const unsigned short* __restrict__ hT,
                                                   float* __restrict__ out) {
    __shared__ unsigned short smemB[128 * D_DIM];   // 32 KB b-strip

    const int tid  = threadIdx.x;
    const int lane = tid & 31;
    const int wave = tid >> 5;
    const int r    = lane & 15;
    const int hi   = lane >> 4;

    const int p0    = (blockIdx.x * 8 + wave) * 16;   // this wave's p-tile
    const int bbase = blockIdx.y * 128;               // block's b strip

    // ---- Async-stage B strip into LDS: 2048 x 16B chunks, 8 per thread ----
    {
        const char*  gsrc    = (const char*)(hT + (size_t)bbase * D_DIM);
        unsigned     ldsBase = (unsigned)(uintptr_t)(&smemB[0]);  // low 32b = LDS offset
#pragma unroll
        for (int i = 0; i < 8; ++i) {
            unsigned off = (unsigned)(tid + i * 256) * 16u;
            unsigned long long ga = (unsigned long long)(uintptr_t)(gsrc + off);
            unsigned la = ldsBase + off;
            asm volatile("global_load_async_to_lds_b128 %0, %1, off"
                         :: "v"(la), "v"(ga) : "memory");
        }
        asm volatile("s_wait_asynccnt 0x0" ::: "memory");
    }
    __syncthreads();

    // ---- Load A (V rows p0..p0+15, K=128) into registers, WMMA layout ----
    const unsigned short* arow = Vbf + (size_t)(p0 + r) * D_DIM;
    v16bf a[4];
#pragma unroll
    for (int kc = 0; kc < 4; ++kc) {
        v8bf alo = *(const v8bf*)(arow + kc * 32 + hi * 8);
        v8bf ahi = *(const v8bf*)(arow + kc * 32 + 16 + hi * 8);
        a[kc] = __builtin_shufflevector(alo, ahi,
                                        0, 1, 2, 3, 4, 5, 6, 7,
                                        8, 9, 10, 11, 12, 13, 14, 15);
    }

    // ---- 8 N-tiles, K-chunk outer so a[kc] feeds 8 back-to-back WMMAs ----
    v8f c[8];
#pragma unroll
    for (int t = 0; t < 8; ++t) c[t] = (v8f){0.f,0.f,0.f,0.f,0.f,0.f,0.f,0.f};

#pragma unroll
    for (int kc = 0; kc < 4; ++kc) {
#pragma unroll
        for (int t = 0; t < 8; ++t) {
            const unsigned short* brow = &smemB[(t * 16 + r) * D_DIM + kc * 32 + hi * 8];
            v8bf blo = *(const v8bf*)brow;          // ds_load_b128
            v8bf bhi = *(const v8bf*)(brow + 16);   // ds_load_b128
            v16bf bv = __builtin_shufflevector(blo, bhi,
                                               0, 1, 2, 3, 4, 5, 6, 7,
                                               8, 9, 10, 11, 12, 13, 14, 15);
            c[t] = __builtin_amdgcn_wmma_f32_16x16x32_bf16(
                       false, a[kc], false, bv, (short)0, c[t], false, false);
        }
    }

    // ---- Non-temporal stores: lane writes out[b0+r][p0+hi*8 .. +7] ----
#pragma unroll
    for (int t = 0; t < 8; ++t) {
        const int b0 = bbase + t * 16;
        float* orow = out + (size_t)(b0 + r) * P_DIM + p0 + hi * 8;
        v4f q0 = {c[t][0], c[t][1], c[t][2], c[t][3]};
        v4f q1 = {c[t][4], c[t][5], c[t][6], c[t][7]};
        __builtin_nontemporal_store(q0, (v4f*)orow);
        __builtin_nontemporal_store(q1, (v4f*)(orow + 4));
    }
}

// ---------------------------------------------------------------------------
// Launcher
// inputs: d_in[0]=x int32 [B,2], d_in[1]=W f32 [D,P], d_in[2]=V f32 [P,D]
// d_out = f32 [B,P]; workspace: Vbf (8,192,000 B) | hiddenT (1,048,576 B)
// ---------------------------------------------------------------------------
extern "C" void kernel_launch(void* const* d_in, const int* in_sizes, int n_in,
                              void* d_out, int out_size, void* d_ws, size_t ws_size,
                              hipStream_t stream) {
    const int*   x = (const int*)d_in[0];
    const float* W = (const float*)d_in[1];
    const float* V = (const float*)d_in[2];
    float*     out = (float*)d_out;

    unsigned short* Vbf = (unsigned short*)d_ws;
    unsigned short* hT  = (unsigned short*)((char*)d_ws +
                           (size_t)P_DIM * D_DIM * sizeof(unsigned short));

    // 1) V -> bf16 : 1,024,000 threads -> 4000 blocks x 256 x4 (exact)
    k_convert_V<<<dim3((P_DIM * D_DIM) / (256 * 4)), dim3(256), 0, stream>>>(V, Vbf);

    // 2) hiddenT : B*D = 524,288 -> 2048 blocks x 256 (exact)
    k_build_hidden<<<dim3((B_DIM * D_DIM) / 256), dim3(256), 0, stream>>>(x, W, hT);

    // 3) WMMA GEMM: grid (2000/8, 4096/128) = (250, 32), 8 waves/block
    k_gemm_wmma<<<dim3(250, 32), dim3(256), 0, stream>>>(Vbf, hT, out);
}
